// FastHybridModel_33784212750508
// MI455X (gfx1250) — matrix operand order repaired and checked by reading, not compiled, not run on hardware
//
#include <hip/hip_runtime.h>
#include <hip/hip_bf16.h>
#include <cstddef>

// ---------------------------------------------------------------------------
// Problem constants (from reference)
// ---------------------------------------------------------------------------
#define BATCH   8192
#define LIN     1024
#define TSTEPS  32
#define HID     32
#define NTILES  (BATCH / 16)            // 512 sixteen-row batch tiles
#define SP_OFF  (BATCH * 4)             // spatial region in d_out
#define HL_OFF  (BATCH * 4 + BATCH * 32) // h_last region in d_out
#define EPSBN   1e-5f

typedef __attribute__((ext_vector_type(16))) _Float16 v16h;
typedef __attribute__((ext_vector_type(8)))  _Float16 v8h;
typedef __attribute__((ext_vector_type(8)))  float    v8f;

static __device__ __forceinline__ v8f wmma_f16(v16h a, v16h b, v8f c) {
  // D = A(16x32 f16) * B(32x16 f16) + C(16x16 f32)
  return __builtin_amdgcn_wmma_f32_16x16x32_f16(false, a, false, b,
                                                (short)0, c, false, false);
}

static __device__ __forceinline__ v8f splat8(float x) {
  v8f r;
#pragma unroll
  for (int i = 0; i < 8; ++i) r[i] = x;
  return r;
}

static __device__ __forceinline__ float sigm(float x) {
  return 1.0f / (1.0f + __expf(-x));
}
static __device__ __forceinline__ float tanh_fast(float x) {
  x = fminf(fmaxf(x, -20.0f), 20.0f);
  float e = __expf(2.0f * x);
  return (e - 1.0f) / (e + 1.0f);
}

// ===========================================================================
// Kernel 1: fused conv stack (conv+bias+BN+ReLU x3) + spatial mean + LIF scan.
// One block (256 threads) per batch row; all intermediates in LDS.
// Spikes are written to workspace already swizzled into the CDNA5 16-bit
// A-matrix register image: spk[tile][t][lane][16 halfs] where for feature k:
//   lane = m + 16*((k>>3)&1),  half = (k&7) | ((k>>4)<<3)      (m = b % 16)
// ===========================================================================
__global__ __launch_bounds__(256) void k1_conv_lif(
    const float* __restrict__ x,
    const float* __restrict__ w1, const float* __restrict__ b1,
    const float* __restrict__ g1, const float* __restrict__ be1,
    const float* __restrict__ m1, const float* __restrict__ v1,
    const float* __restrict__ w2, const float* __restrict__ b2,
    const float* __restrict__ g2, const float* __restrict__ be2,
    const float* __restrict__ m2, const float* __restrict__ v2,
    const float* __restrict__ w3, const float* __restrict__ b3,
    const float* __restrict__ g3, const float* __restrict__ be3,
    const float* __restrict__ m3, const float* __restrict__ v3,
    _Float16* __restrict__ spk, float* __restrict__ out)
{
  __shared__ float xs[LIN];
  __shared__ float a1[8][256];
  __shared__ float a2[16][64];
  __shared__ float a3[32][32];
  __shared__ float w1s[8][7];
  __shared__ float w2s[16][8][5];
  __shared__ float w3s[32][16][3];
  __shared__ float s1[8],  t1[8];
  __shared__ float s2[16], t2[16];
  __shared__ float s3[32], t3[32];

  const int tid = threadIdx.x;
  const int b   = blockIdx.x;

  if (tid == 0) __builtin_prefetch(&x[(size_t)b * LIN], 0, 3);

  // --- stage input + weights + folded BN params into LDS -------------------
  {
    const float4* x4 = (const float4*)(x + (size_t)b * LIN);
    ((float4*)xs)[tid] = x4[tid];
  }
  if (tid < 56) { int c = tid / 7, k = tid % 7; w1s[c][k] = w1[tid]; }
  for (int idx = tid; idx < 640; idx += 256) {
    int o = idx / 40, r = idx % 40;
    w2s[o][r / 5][r % 5] = w2[idx];
  }
  for (int idx = tid; idx < 1536; idx += 256) {
    int o = idx / 48, r = idx % 48;
    w3s[o][r / 3][r % 3] = w3[idx];
  }
  if (tid < 8) {
    float sc = g1[tid] * __frsqrt_rn(v1[tid] + EPSBN);
    s1[tid] = sc; t1[tid] = b1[tid] * sc + (be1[tid] - m1[tid] * sc);
  }
  if (tid < 16) {
    float sc = g2[tid] * __frsqrt_rn(v2[tid] + EPSBN);
    s2[tid] = sc; t2[tid] = b2[tid] * sc + (be2[tid] - m2[tid] * sc);
  }
  if (tid < 32) {
    float sc = g3[tid] * __frsqrt_rn(v3[tid] + EPSBN);
    s3[tid] = sc; t3[tid] = b3[tid] * sc + (be3[tid] - m3[tid] * sc);
  }
  __syncthreads();

  // --- conv1: 1->8 ch, k=7, stride=4, pad=3 : 8x256 outputs ----------------
#pragma unroll
  for (int j = 0; j < 8; ++j) {
    int idx = tid + j * 256;
    int c = idx >> 8, p = idx & 255;
    int l0 = p * 4 - 3;
    float s = 0.f;
#pragma unroll
    for (int k = 0; k < 7; ++k) {
      int l = l0 + k;
      if (l >= 0 && l < LIN) s += xs[l] * w1s[c][k];
    }
    a1[c][p] = fmaxf(s * s1[c] + t1[c], 0.f);
  }
  __syncthreads();

  // --- conv2: 8->16 ch, k=5, stride=4, pad=2 : 16x64 outputs ---------------
#pragma unroll
  for (int j = 0; j < 4; ++j) {
    int idx = tid + j * 256;
    int c = idx >> 6, p = idx & 63;
    int l0 = p * 4 - 2;
    float s = 0.f;
#pragma unroll
    for (int ci = 0; ci < 8; ++ci)
#pragma unroll
      for (int k = 0; k < 5; ++k) {
        int l = l0 + k;
        if (l >= 0 && l < 256) s += a1[ci][l] * w2s[c][ci][k];
      }
    a2[c][p] = fmaxf(s * s2[c] + t2[c], 0.f);
  }
  __syncthreads();

  // --- conv3: 16->32 ch, k=3, stride=2, pad=1 : 32x32 outputs --------------
#pragma unroll
  for (int j = 0; j < 4; ++j) {
    int idx = tid + j * 256;
    int c = idx >> 5, p = idx & 31;
    int l0 = p * 2 - 1;
    float s = 0.f;
#pragma unroll
    for (int ci = 0; ci < 16; ++ci)
#pragma unroll
      for (int k = 0; k < 3; ++k) {
        int l = l0 + k;
        if (l >= 0 && l < 64) s += a2[ci][l] * w3s[c][ci][k];
      }
    a3[c][p] = fmaxf(s * s3[c] + t3[c], 0.f);
  }
  __syncthreads();

  // --- spatial mean + LIF spike scan (per feature channel) -----------------
  if (tid < 32) {
    const int f = tid;
    float acc = 0.f;
#pragma unroll
    for (int p = 0; p < 32; ++p) acc += a3[f][p];
    out[SP_OFF + b * 32 + f] = acc * (1.0f / 32.0f);

    // LIF: mem = 0.9*mem + in; spike = mem > 0.5; mem *= (1-spike)
    const int tile = b >> 4, m = b & 15;
    const int lane = m + (((f >> 3) & 1) << 4);
    const int half = (f & 7) | ((f >> 4) << 3);
    float mem = 0.f;
    for (int t = 0; t < TSTEPS; ++t) {
      mem = 0.9f * mem + a3[f][t];            // snn_in[b][t][f] = h[b][f][t]
      float spike = (mem > 0.5f) ? 1.0f : 0.0f;
      mem *= (1.0f - spike);
      spk[(((size_t)tile * TSTEPS + t) * 32 + lane) * 16 + half] =
          (_Float16)spike;
    }
  }
}

// ===========================================================================
// Kernel 2: 32-step LSTM via WMMA. One wave per 16-batch tile (8 waves/block).
// Gates[16x128] = 8 N-tiles; per tile: 2 chained v_wmma_f32_16x16x32_f16
// (spike A * w_ih^T  and  h A * w_hh^T), bias preloaded into C.
// i/f/g/o occupy N-tiles {p,2+p,4+p,6+p} -> cell update is lane-local.
// h C/D->A relayout through per-wave LDS scratch + s_wait_dscnt.
// ===========================================================================
__global__ __launch_bounds__(256) void k2_lstm(
    const _Float16* __restrict__ spk,
    const float* __restrict__ w_ih, const float* __restrict__ w_hh,
    const float* __restrict__ b_ih, const float* __restrict__ b_hh,
    float* __restrict__ out)
{
  __shared__ __align__(32) _Float16 wihB[8 * 512];   // [nt][lane][16]
  __shared__ __align__(32) _Float16 whhB[8 * 512];
  __shared__ float biasv[128];
  __shared__ __align__(16) _Float16 hsc[8][16][32];  // per-wave h scratch

  const int tid = threadIdx.x;

  // Build B-matrix images: lane n<16 holds K=0..7,16..23 (2 per VGPR),
  // lane n+16 holds K=8..15,24..31; value = W[n_global][k].
  for (int idx = tid; idx < 4096; idx += 256) {
    int nt = idx >> 9, rem = idx & 511;
    int lane = rem >> 4, hh = rem & 15;
    int n = lane & 15;
    int k = (hh & 7) | ((lane >> 4) << 3) | ((hh >> 3) << 4);
    wihB[idx] = (_Float16)w_ih[(nt * 16 + n) * 32 + k];
    whhB[idx] = (_Float16)w_hh[(nt * 16 + n) * 32 + k];
  }
  if (tid < 128) biasv[tid] = b_ih[tid] + b_hh[tid];
  __syncthreads();

  const int w      = tid >> 5;
  const int lane   = tid & 31;
  const int lanehi = lane >> 4;
  const int n      = lane & 15;
  const int tile   = blockIdx.x * 8 + w;

  float bias_n[8];
#pragma unroll
  for (int q = 0; q < 8; ++q) bias_n[q] = biasv[q * 16 + n];

  v8f  cstate[2] = {splat8(0.f), splat8(0.f)};
  v8f  hstate[2] = {splat8(0.f), splat8(0.f)};
  v16h hA        = {};                       // h == 0 at t=0

  for (int t = 0; t < TSTEPS; ++t) {
    // spike A tile: pre-swizzled register image, 32B per lane
    const v16h xA =
        *(const v16h*)(spk + (((size_t)tile * TSTEPS + t) * 32 + lane) * 16);

#pragma unroll
    for (int p = 0; p < 2; ++p) {
      v8f aI = splat8(bias_n[0 + p]);
      v8f aF = splat8(bias_n[2 + p]);
      v8f aG = splat8(bias_n[4 + p]);
      v8f aO = splat8(bias_n[6 + p]);

      const int lo16 = lane << 4;
      aI = wmma_f16(xA, *(const v16h*)(wihB + ((0 + p) << 9) + lo16), aI);
      aI = wmma_f16(hA, *(const v16h*)(whhB + ((0 + p) << 9) + lo16), aI);
      aF = wmma_f16(xA, *(const v16h*)(wihB + ((2 + p) << 9) + lo16), aF);
      aF = wmma_f16(hA, *(const v16h*)(whhB + ((2 + p) << 9) + lo16), aF);
      aG = wmma_f16(xA, *(const v16h*)(wihB + ((4 + p) << 9) + lo16), aG);
      aG = wmma_f16(hA, *(const v16h*)(whhB + ((4 + p) << 9) + lo16), aG);
      aO = wmma_f16(xA, *(const v16h*)(wihB + ((6 + p) << 9) + lo16), aO);
      aO = wmma_f16(hA, *(const v16h*)(whhB + ((6 + p) << 9) + lo16), aO);

#pragma unroll
      for (int r = 0; r < 8; ++r) {
        float iv = sigm(aI[r]);
        float fv = sigm(aF[r]);
        float gv = tanh_fast(aG[r]);
        float ov = sigm(aO[r]);
        float c  = fv * cstate[p][r] + iv * gv;
        cstate[p][r] = c;
        hstate[p][r] = ov * tanh_fast(c);
      }
      // scatter h (C/D layout) -> LDS [m][hidden]
#pragma unroll
      for (int r = 0; r < 8; ++r) {
        int m = r + (lanehi << 3);
        hsc[w][m][(p << 4) + n] = (_Float16)hstate[p][r];
      }
    }

    // wave-private LDS RAW: wait DS pipe, then gather h into A image
    asm volatile("s_wait_dscnt 0x0" ::: "memory");
    {
      const v8h lo = *(const v8h*)(&hsc[w][n][lanehi * 8]);
      const v8h hi = *(const v8h*)(&hsc[w][n][16 + lanehi * 8]);
#pragma unroll
      for (int i = 0; i < 8; ++i) { hA[i] = lo[i]; hA[8 + i] = hi[i]; }
    }
  }

  // write h_last [B,32] (f32) into d_out region 2
#pragma unroll
  for (int p = 0; p < 2; ++p)
#pragma unroll
    for (int r = 0; r < 8; ++r) {
      int m = r + (lanehi << 3);
      int b = tile * 16 + m;
      out[HL_OFF + b * 32 + (p << 4) + n] = hstate[p][r];
    }
}

// ===========================================================================
// Kernel 3: FC head via WMMA. combined[16x64] = [h_last | spatial].
// fc1 (64->64): K=64 as two chained 16x16x32 WMMAs per N-tile (4 tiles).
// fc2 (64->32): 2 N-tiles.  fc3 (32->4): one N-tile padded with zero cols.
// ===========================================================================
__global__ __launch_bounds__(256) void k3_fc(
    const float* __restrict__ fc1w, const float* __restrict__ fc1b,
    const float* __restrict__ fc2w, const float* __restrict__ fc2b,
    const float* __restrict__ fc3w, const float* __restrict__ fc3b,
    float* __restrict__ out)
{
  __shared__ __align__(32) _Float16 f1B[2 * 4 * 512];  // [q][nt][lane][16]
  __shared__ __align__(32) _Float16 f2B[2 * 2 * 512];
  __shared__ __align__(32) _Float16 f3B[512];
  __shared__ __align__(16) _Float16 zsc[8][16][64];

  const int tid = threadIdx.x;

  for (int idx = tid; idx < 4096; idx += 256) {
    int q = idx >> 11, nt = (idx >> 9) & 3;
    int lane = (idx >> 4) & 31, hh = idx & 15;
    int n = lane & 15;
    int k = (q << 5) + ((lane >> 4) << 3) + (hh & 7) + ((hh >> 3) << 4);
    f1B[idx] = (_Float16)fc1w[(nt * 16 + n) * 64 + k];
  }
  for (int idx = tid; idx < 2048; idx += 256) {
    int q = idx >> 10, nt = (idx >> 9) & 1;
    int lane = (idx >> 4) & 31, hh = idx & 15;
    int n = lane & 15;
    int k = (q << 5) + ((lane >> 4) << 3) + (hh & 7) + ((hh >> 3) << 4);
    f2B[idx] = (_Float16)fc2w[(nt * 16 + n) * 64 + k];
  }
  for (int idx = tid; idx < 512; idx += 256) {
    int lane = idx >> 4, hh = idx & 15;
    int n = lane & 15;
    int k = ((lane >> 4) << 3) + (hh & 7) + ((hh >> 3) << 4);
    f3B[idx] = (n < 4) ? (_Float16)fc3w[n * 32 + k] : (_Float16)0.0f;
  }
  __syncthreads();

  const int w      = tid >> 5;
  const int lane   = tid & 31;
  const int lanehi = lane >> 4;
  const int n      = lane & 15;
  const int tile   = blockIdx.x * 8 + w;
  const int m      = n;          // A-image row held by this lane
  const int b_row  = tile * 16 + m;

  // combined A image: k 0..31 = h_last, k 32..63 = spatial
  v16h A0, A1;
#pragma unroll
  for (int hh = 0; hh < 16; ++hh) {
    int k = (hh & 7) | (lanehi << 3) | ((hh >> 3) << 4);   // 0..31
    A0[hh] = (_Float16)out[HL_OFF + b_row * 32 + k];
    A1[hh] = (_Float16)out[SP_OFF + b_row * 32 + k];
  }

  // ---- fc1: relu(combined @ fc1w.T + b) ----
#pragma unroll
  for (int nt = 0; nt < 4; ++nt) {
    v8f a = splat8(fc1b[nt * 16 + n]);
    a = wmma_f16(A0, *(const v16h*)(f1B + ((0 * 4 + nt) << 9) + (lane << 4)), a);
    a = wmma_f16(A1, *(const v16h*)(f1B + ((1 * 4 + nt) << 9) + (lane << 4)), a);
#pragma unroll
    for (int r = 0; r < 8; ++r) {
      int mr = r + (lanehi << 3);
      zsc[w][mr][nt * 16 + n] = (_Float16)fmaxf(a[r], 0.f);
    }
  }
  asm volatile("s_wait_dscnt 0x0" ::: "memory");

  v16h zA0, zA1;
  {
    const v8h l0 = *(const v8h*)(&zsc[w][m][lanehi * 8]);
    const v8h h0 = *(const v8h*)(&zsc[w][m][16 + lanehi * 8]);
    const v8h l1 = *(const v8h*)(&zsc[w][m][32 + lanehi * 8]);
    const v8h h1 = *(const v8h*)(&zsc[w][m][48 + lanehi * 8]);
#pragma unroll
    for (int i = 0; i < 8; ++i) {
      zA0[i] = l0[i]; zA0[8 + i] = h0[i];
      zA1[i] = l1[i]; zA1[8 + i] = h1[i];
    }
  }

  // ---- fc2: relu(z @ fc2w.T + b) ----
#pragma unroll
  for (int nt = 0; nt < 2; ++nt) {
    v8f a = splat8(fc2b[nt * 16 + n]);
    a = wmma_f16(zA0, *(const v16h*)(f2B + ((0 * 2 + nt) << 9) + (lane << 4)), a);
    a = wmma_f16(zA1, *(const v16h*)(f2B + ((1 * 2 + nt) << 9) + (lane << 4)), a);
#pragma unroll
    for (int r = 0; r < 8; ++r) {
      int mr = r + (lanehi << 3);
      zsc[w][mr][nt * 16 + n] = (_Float16)fmaxf(a[r], 0.f);
    }
  }
  asm volatile("s_wait_dscnt 0x0" ::: "memory");

  v16h zA;
  {
    const v8h lo = *(const v8h*)(&zsc[w][m][lanehi * 8]);
    const v8h hi = *(const v8h*)(&zsc[w][m][16 + lanehi * 8]);
#pragma unroll
    for (int i = 0; i < 8; ++i) { zA[i] = lo[i]; zA[8 + i] = hi[i]; }
  }

  // ---- fc3: z @ fc3w.T + b  (N=4 padded into one 16-wide tile) ----
  v8f acc = splat8(0.f);
  acc = wmma_f16(zA, *(const v16h*)(f3B + (lane << 4)), acc);

  if (n < 4) {
    float bb = fc3b[n];
#pragma unroll
    for (int r = 0; r < 8; ++r) {
      int mr = r + (lanehi << 3);
      out[(tile * 16 + mr) * 4 + n] = acc[r] + bb;
    }
  }
}

// ===========================================================================
extern "C" void kernel_launch(void* const* d_in, const int* in_sizes, int n_in,
                              void* d_out, int out_size, void* d_ws,
                              size_t ws_size, hipStream_t stream)
{
  const float* x    = (const float*)d_in[0];
  const float* w1   = (const float*)d_in[1];
  const float* b1   = (const float*)d_in[2];
  const float* g1   = (const float*)d_in[3];
  const float* be1  = (const float*)d_in[4];
  const float* m1   = (const float*)d_in[5];
  const float* v1   = (const float*)d_in[6];
  const float* w2   = (const float*)d_in[7];
  const float* b2   = (const float*)d_in[8];
  const float* g2   = (const float*)d_in[9];
  const float* be2  = (const float*)d_in[10];
  const float* m2   = (const float*)d_in[11];
  const float* v2   = (const float*)d_in[12];
  const float* w3   = (const float*)d_in[13];
  const float* b3   = (const float*)d_in[14];
  const float* g3   = (const float*)d_in[15];
  const float* be3  = (const float*)d_in[16];
  const float* m3   = (const float*)d_in[17];
  const float* v3   = (const float*)d_in[18];
  const float* w_ih = (const float*)d_in[19];
  const float* w_hh = (const float*)d_in[20];
  const float* b_ih = (const float*)d_in[21];
  const float* b_hh = (const float*)d_in[22];
  const float* fc1w = (const float*)d_in[23];
  const float* fc1b = (const float*)d_in[24];
  const float* fc2w = (const float*)d_in[25];
  const float* fc2b = (const float*)d_in[26];
  const float* fc3w = (const float*)d_in[27];
  const float* fc3b = (const float*)d_in[28];

  float*     out = (float*)d_out;
  _Float16*  spk = (_Float16*)d_ws;   // 512*32*32*16 halfs = 16 MB A-images

  k1_conv_lif<<<BATCH, 256, 0, stream>>>(
      x, w1, b1, g1, be1, m1, v1, w2, b2, g2, be2, m2, v2,
      w3, b3, g3, be3, m3, v3, spk, out);

  k2_lstm<<<NTILES / 8, 256, 0, stream>>>(spk, w_ih, w_hh, b_ih, b_hh, out);

  k3_fc<<<NTILES / 8, 256, 0, stream>>>(fc1w, fc1b, fc2w, fc2b,
                                        fc3w, fc3b, out);
}
